// MLP_Q_84043920048757
// MI455X (gfx1250) — 8-bit matrix operand order repaired and checked by reading, compiled, not run on hardware
//
#include <hip/hip_runtime.h>
#include <math.h>

typedef int v8i __attribute__((ext_vector_type(8)));

#define BATCH 4096
#define DIN   1024
#define HID   4096
#define DOUT  1024

#define OFF_U      0
#define OFF_SC     64
#define OFF_XQ     256
#define OFF_W1T    (OFF_XQ  + (size_t)BATCH * DIN)
#define OFF_W2T    (OFF_W1T + (size_t)HID * DIN)
#define OFF_B1Q    (OFF_W2T + (size_t)DOUT * HID)
#define OFF_B2Q    (OFF_B1Q + 4096)
#define OFF_ALO    (OFF_B2Q + 4096)
#define OFF_AHI    (OFF_ALO + (size_t)BATCH * HID)

__global__ void mlpq_init_kernel(unsigned* u) {
    if (threadIdx.x < 3) u[threadIdx.x] = 0u;
}

__global__ void mlpq_absmax2_kernel(const float* __restrict__ a, long na,
                                    const float* __restrict__ b, long nb,
                                    unsigned* __restrict__ out) {
    long i = (long)blockIdx.x * blockDim.x + threadIdx.x;
    long stride = (long)gridDim.x * blockDim.x;
    float m = 0.0f;
    for (long j = i; j < na; j += stride) m = fmaxf(m, fabsf(a[j]));
    if (b) for (long j = i; j < nb; j += stride) m = fmaxf(m, fabsf(b[j]));
    for (int off = 16; off > 0; off >>= 1)
        m = fmaxf(m, __shfl_down(m, off, 32));
    if ((threadIdx.x & 31) == 0)
        atomicMax(out, __float_as_uint(m));
}

__global__ void mlpq_finalize_scales_kernel(const unsigned* __restrict__ u,
                                            float* __restrict__ sc) {
    if (threadIdx.x == 0) {
        sc[0] = __uint_as_float(u[0]) / 127.0f;
        sc[1] = __uint_as_float(u[1]) / 127.0f;
        sc[2] = __uint_as_float(u[2]) / 127.0f;
    }
}

__global__ void mlpq_quant_vec_kernel(const float* __restrict__ src,
                                      signed char* __restrict__ dst,
                                      const float* __restrict__ sc, int scIdx,
                                      long n) {
    float inv = 1.0f / sc[scIdx];
    long i = (long)blockIdx.x * blockDim.x + threadIdx.x;
    long stride = (long)gridDim.x * blockDim.x;
    for (long j = i; j < n; j += stride)
        dst[j] = (signed char)(int)rintf(src[j] * inv);
}

__global__ void mlpq_quant_wt_kernel(const float* __restrict__ w,
                                     signed char* __restrict__ wt,
                                     const float* __restrict__ sc, int scIdx,
                                     int K, int N) {
    float inv = 1.0f / sc[scIdx];
    long total = (long)K * N;
    long i = (long)blockIdx.x * blockDim.x + threadIdx.x;
    long stride = (long)gridDim.x * blockDim.x;
    for (long j = i; j < total; j += stride) {
        int k = (int)(j / N);
        int n = (int)(j - (long)k * N);
        wt[(long)n * K + k] = (signed char)(int)rintf(w[j] * inv);
    }
}

__device__ __forceinline__ v8i mlpq_load_a_frag(const signed char* __restrict__ tile,
                                                int ld, int lane) {
    int row  = lane & 15;
    int koff = (lane & 16) ? 16 : 0;
    const signed char* p = tile + (long)row * ld + koff;
    int4 c0 = *(const int4*)(p);
    int4 c1 = *(const int4*)(p + 32);
    v8i a;
    a[0] = c0.x; a[1] = c0.y; a[2] = c0.z; a[3] = c0.w;
    a[4] = c1.x; a[5] = c1.y; a[6] = c1.z; a[7] = c1.w;
    return a;
}

__device__ __forceinline__ v8i mlpq_load_b_frag(const signed char* __restrict__ tile,
                                                int ld, int lane) {
    int col  = lane & 15;
    int koff = (lane & 16) ? 16 : 0;
    const signed char* p = tile + (long)col * ld + koff;
    int4 c0 = *(const int4*)(p);
    int4 c1 = *(const int4*)(p + 32);
    v8i b;
    b[0] = c0.x; b[1] = c0.y; b[2] = c0.z; b[3] = c0.w;
    b[4] = c1.x; b[5] = c1.y; b[6] = c1.z; b[7] = c1.w;
    return b;
}

__global__ void mlpq_gemm1_kernel(const signed char* __restrict__ xq,
                                  const signed char* __restrict__ w1t,
                                  const signed char* __restrict__ b1q,
                                  const float* __restrict__ sc,
                                  unsigned char* __restrict__ act_lo,
                                  unsigned char* __restrict__ act_hi) {
    int lane = threadIdx.x & 31;
    int wave = threadIdx.x >> 5;
    int tile = blockIdx.x * 8 + wave;
    const int TN = HID / 64;
    int m0 = (tile / TN) * 32;
    int n0 = (tile % TN) * 64;

    float xs = sc[2];
    v8i acc[2][4];
#pragma unroll
    for (int i = 0; i < 2; ++i)
#pragma unroll
        for (int j = 0; j < 4; ++j) acc[i][j] = (v8i){0,0,0,0,0,0,0,0};

    const signed char* aBase = xq  + (long)m0 * DIN;
    const signed char* bBase = w1t + (long)n0 * DIN;

    for (int k0 = 0; k0 < DIN; k0 += 64) {
        if (k0 + 64 < DIN)
            __builtin_prefetch(bBase + (long)(lane & 15) * DIN + k0 + 64, 0, 1);
        v8i a0 = mlpq_load_a_frag(aBase + k0,                 DIN, lane);
        v8i a1 = mlpq_load_a_frag(aBase + (long)16 * DIN + k0, DIN, lane);
        v8i b[4];
#pragma unroll
        for (int j = 0; j < 4; ++j)
            b[j] = mlpq_load_b_frag(bBase + (long)(16 * j) * DIN + k0, DIN, lane);
#pragma unroll
        for (int j = 0; j < 4; ++j) {
            acc[0][j] = __builtin_amdgcn_wmma_i32_16x16x64_iu8(true, a0, true, b[j],
                                                               acc[0][j], false, false);
            acc[1][j] = __builtin_amdgcn_wmma_i32_16x16x64_iu8(true, a1, true, b[j],
                                                               acc[1][j], false, false);
        }
    }

    int n  = lane & 15;
    int mb = (lane & 16) ? 8 : 0;
#pragma unroll
    for (int i = 0; i < 2; ++i) {
#pragma unroll
        for (int j = 0; j < 4; ++j) {
            int bq = (int)b1q[n0 + 16 * j + n];
            long rowBase = (long)(m0 + 16 * i + mb);
            for (int r = 0; r < 8; ++r) {
                short o1 = (short)((int)rintf((float)acc[i][j][r] * xs) + bq);
                int relu = (o1 > 0) ? (int)o1 : 0;
                long idx = (rowBase + r) * HID + (n0 + 16 * j + n);
                act_lo[idx] = (unsigned char)(relu & 0xFF);
                act_hi[idx] = (unsigned char)(relu >> 8);
            }
        }
    }
}

__global__ void mlpq_gemm2_kernel(const unsigned char* __restrict__ act_lo,
                                  const unsigned char* __restrict__ act_hi,
                                  const signed char* __restrict__ w2t,
                                  const signed char* __restrict__ b2q,
                                  const float* __restrict__ sc,
                                  float* __restrict__ out) {
    int lane = threadIdx.x & 31;
    int wave = threadIdx.x >> 5;
    int tile = blockIdx.x * 8 + wave;
    const int TN = DOUT / 64;
    int m0 = (tile / TN) * 16;
    int n0 = (tile % TN) * 64;

    float s1 = sc[0];
    float s2 = sc[1];

    v8i accL[4], accH[4];
#pragma unroll
    for (int j = 0; j < 4; ++j) {
        accL[j] = (v8i){0,0,0,0,0,0,0,0};
        accH[j] = (v8i){0,0,0,0,0,0,0,0};
    }

    const signed char* loBase = (const signed char*)act_lo + (long)m0 * HID;
    const signed char* hiBase = (const signed char*)act_hi + (long)m0 * HID;
    const signed char* bBase  = w2t + (long)n0 * HID;

#pragma unroll 4
    for (int k0 = 0; k0 < HID; k0 += 64) {
        if (k0 + 64 < HID)
            __builtin_prefetch(bBase + (long)(lane & 15) * HID + k0 + 64, 0, 1);
        v8i al = mlpq_load_a_frag(loBase + k0, HID, lane);
        v8i ah = mlpq_load_a_frag(hiBase + k0, HID, lane);
        v8i b[4];
#pragma unroll
        for (int j = 0; j < 4; ++j)
            b[j] = mlpq_load_b_frag(bBase + (long)(16 * j) * HID + k0, HID, lane);
#pragma unroll
        for (int j = 0; j < 4; ++j) {
            accL[j] = __builtin_amdgcn_wmma_i32_16x16x64_iu8(false, al, true, b[j],
                                                             accL[j], false, false);
            accH[j] = __builtin_amdgcn_wmma_i32_16x16x64_iu8(false, ah, true, b[j],
                                                             accH[j], false, false);
        }
    }

    int n  = lane & 15;
    int mb = (lane & 16) ? 8 : 0;
#pragma unroll
    for (int j = 0; j < 4; ++j) {
        int bq = (int)b2q[n0 + 16 * j + n];
        for (int r = 0; r < 8; ++r) {
            int y0 = accL[j][r] + (accH[j][r] << 8);
            short o2 = (short)((int)rintf((float)y0 * s1) + bq);
            out[(long)(m0 + mb + r) * DOUT + (n0 + 16 * j + n)] = (float)o2 * s2;
        }
    }
}

extern "C" void kernel_launch(void* const* d_in, const int* in_sizes, int n_in,
                              void* d_out, int out_size, void* d_ws, size_t ws_size,
                              hipStream_t stream) {
    const float* x   = (const float*)d_in[0];
    const float* w1f = (const float*)d_in[1];
    const float* b1f = (const float*)d_in[2];
    const float* w2f = (const float*)d_in[3];
    const float* b2f = (const float*)d_in[4];
    float* outp = (float*)d_out;

    char* ws = (char*)d_ws;
    unsigned*      u     = (unsigned*)(ws + OFF_U);
    float*         sc    = (float*)(ws + OFF_SC);
    signed char*   xq    = (signed char*)(ws + OFF_XQ);
    signed char*   w1t   = (signed char*)(ws + OFF_W1T);
    signed char*   w2t   = (signed char*)(ws + OFF_W2T);
    signed char*   b1q   = (signed char*)(ws + OFF_B1Q);
    signed char*   b2q   = (signed char*)(ws + OFF_B2Q);
    unsigned char* actLo = (unsigned char*)(ws + OFF_ALO);
    unsigned char* actHi = (unsigned char*)(ws + OFF_AHI);

    mlpq_init_kernel<<<1, 32, 0, stream>>>(u);
    mlpq_absmax2_kernel<<<2048, 256, 0, stream>>>(w1f, (long)DIN * HID, b1f, (long)HID, u + 0);
    mlpq_absmax2_kernel<<<2048, 256, 0, stream>>>(w2f, (long)HID * DOUT, b2f, (long)DOUT, u + 1);
    mlpq_absmax2_kernel<<<2048, 256, 0, stream>>>(x,   (long)BATCH * DIN, (const float*)nullptr, 0, u + 2);
    mlpq_finalize_scales_kernel<<<1, 32, 0, stream>>>(u, sc);

    mlpq_quant_vec_kernel<<<2048, 256, 0, stream>>>(x, xq, sc, 2, (long)BATCH * DIN);
    mlpq_quant_wt_kernel<<<2048, 256, 0, stream>>>(w1f, w1t, sc, 0, DIN, HID);
    mlpq_quant_vec_kernel<<<16, 256, 0, stream>>>(b1f, b1q, sc, 0, (long)HID);
    mlpq_quant_wt_kernel<<<2048, 256, 0, stream>>>(w2f, w2t, sc, 1, HID, DOUT);
    mlpq_quant_vec_kernel<<<4, 256, 0, stream>>>(b2f, b2q, sc, 1, (long)DOUT);

    mlpq_gemm1_kernel<<<8192 / 8, 256, 0, stream>>>(xq, w1t, b1q, sc, actLo, actHi);

    mlpq_gemm2_kernel<<<4096 / 8, 256, 0, stream>>>(actLo, actHi, w2t, b2q, sc, outp);
}
